// LiftedConv_53549652247381
// MI455X (gfx1250) — matrix operand order repaired and checked by reading, not compiled, not run on hardware
//
#include <hip/hip_runtime.h>

typedef __attribute__((ext_vector_type(16))) _Float16 v16h;
typedef __attribute__((ext_vector_type(8)))  float    v8f;

#define TWO_PI_F 6.2831853071795864769f
#define PI_F     3.1415926535897932385f

// ---------------- math helpers (match jnp semantics) ----------------
__device__ __forceinline__ float signf(float t){ return (t > 0.f) ? 1.f : ((t < 0.f) ? -1.f : 0.f); }
__device__ __forceinline__ float pmodf(float a, float m){ return a - floorf(a / m) * m; }
__device__ __forceinline__ float B2f(float x){
  float t1 = -3.f * (x - 0.5f) * (x - 0.5f) * signf(0.5f - x);
  float t2 =        (x - 1.5f) * (x - 1.5f) * signf(1.5f - x);
  float t3 = -0.75f * (1.f + 2.f * x) * (1.f + 2.f * x) * signf(0.5f + x);
  float t4 =  0.25f * (3.f + 2.f * x) * (3.f + 2.f * x) * signf(1.5f + x);
  return 0.25f * (t1 + t2 + t3 + t4);
}

// ---------------- phase 0: X (B,128,H,W) f32 -> Xt (B,H,W,128) f16 ----------------
__global__ void k_xpose(const float* __restrict__ X, _Float16* __restrict__ Xt){
  __shared__ float tile[128][33];
  int b = blockIdx.z, y = blockIdx.y, x0 = blockIdx.x * 32;
  int tx = threadIdx.x & 31, tg = threadIdx.x >> 5;
  for (int ci = tg; ci < 128; ci += 8)
    tile[ci][tx] = X[(((size_t)(b * 128 + ci)) * 128 + y) * 128 + x0 + tx];
  __syncthreads();
  int xx = threadIdx.x >> 3, cg = threadIdx.x & 7;   // xx: 0..31, cg: ci group of 16
  _Float16 ov[16];
  #pragma unroll
  for (int k = 0; k < 16; ++k) ov[k] = (_Float16)tile[cg * 16 + k][xx];
  _Float16* dst = Xt + (((size_t)(b * 128 + y)) * 128 + x0 + xx) * 128 + cg * 16;
  ((uint4*)dst)[0] = ((uint4*)ov)[0];
  ((uint4*)dst)[1] = ((uint4*)ov)[1];
}

// ---------------- phase 1: build filter bank in WMMA B-fragment layout ----------------
// gW layout: [tap(25)][kc(4)][nc(8)][lane(32)][e(16)] f16, 819200 bytes total
// B-frag layout used here (contiguous k, same order as A): lane = kh*16 + n_local, element e <-> K = kh*16 + e
__global__ void k_weights(const float* __restrict__ wts, _Float16* __restrict__ gW){
  int h = blockIdx.x >> 4;      // 0..7  (output h index -> nc block)
  int c = blockIdx.x & 15;      // 0..15 (input c index)
  __shared__ float HH[64];       // [n][b]
  __shared__ float XX[625];      // [ij][p]
  __shared__ float Wc[3200];     // weights[c] : [m=200][o=16]
  __shared__ float T[3200];      // [n][p][o]
  int tid = threadIdx.x;
  float hv = (TWO_PI_F / 8.f) * (float)h;

  for (int idx = tid; idx < 64; idx += 256){
    int n = idx >> 3, bb = idx & 7;
    float hs = pmodf((TWO_PI_F / 8.f) * (float)n - hv, TWO_PI_F);
    float d  = fabsf(pmodf(hs - (TWO_PI_F / 8.f) * (float)bb + PI_F, TWO_PI_F) - PI_F);
    HH[idx] = B2f(d / (TWO_PI_F / 8.f));
  }
  float cs = cosf(-hv), sn = sinf(-hv);
  for (int idx = tid; idx < 625; idx += 256){
    int ij = idx / 25, p = idx - ij * 25;
    float iv = (float)(ij / 5 - 2), jv = (float)(ij % 5 - 2);
    float xs0 = cs * iv - sn * jv;
    float xs1 = sn * iv + cs * jv;
    float d0 = (float)(p / 5 - 2) - xs0;
    float d1 = (float)(p % 5 - 2) - xs1;
    XX[idx] = B2f(d0) * B2f(d1);
  }
  for (int idx = tid; idx < 3200; idx += 256) Wc[idx] = wts[(size_t)c * 3200 + idx];
  __syncthreads();
  for (int idx = tid; idx < 3200; idx += 256){
    int n = idx / 400, p = (idx / 16) % 25, o = idx & 15;
    float s = 0.f;
    #pragma unroll
    for (int bb = 0; bb < 8; ++bb) s += HH[n * 8 + bb] * Wc[(p * 8 + bb) * 16 + o];
    T[idx] = s;
  }
  __syncthreads();
  for (int idx = tid; idx < 3200; idx += 256){
    int o = idx / 200, n = (idx / 25) & 7, ij = idx % 25;
    float s = 0.f;
    #pragma unroll
    for (int p = 0; p < 25; ++p) s += XX[ij * 25 + p] * T[n * 400 + p * 16 + o];
    int ci = c * 8 + n;
    int kc = ci >> 5, kl = ci & 31, kh = kl >> 4, e = kl & 15;
    int lane = kh * 16 + o;                 // n_local = o, nc = h
    size_t off = ((((size_t)ij * 4 + kc) * 8 + h) * 32 + lane) * 16 + e;
    gW[off] = (_Float16)s;
  }
}

// ---------------- phase 2: implicit-GEMM conv, f16 WMMA, f32 accumulate ----------------
// LDS holds only half the ci range (64) -> 95,040 B -> 2 workgroups / WGP (16 waves).
// B fragments are loaded per-wave directly from L2-resident gW (no barriers in main loop).
#define XPITCH 72                      // halves per pixel (64 ci + 8 pad)
#define XROW   (132 * XPITCH)          // 9504 halves per LDS row slot (x in 0..131 incl. halo)
#define SMEM_BYTES (5 * XROW * 2)      // 95040 B dynamic LDS

__global__ void __launch_bounds__(256)
k_conv(const _Float16* __restrict__ Xt, const _Float16* __restrict__ gW,
       float* __restrict__ out){
  extern __shared__ char smem[];
  _Float16* sX = (_Float16*)smem;
  int y = blockIdx.x, b = blockIdx.y;
  int tid  = threadIdx.x;
  int lane = tid & 31, wave = tid >> 5;
  int mw = wave & 1, nw = wave >> 1;   // 2 M-waves x 4 N-waves
  int x0w = mw * 64;
  int lm = lane & 15, kh = lane >> 4;

  v8f acc[4][2];
  #pragma unroll
  for (int mf = 0; mf < 4; ++mf){
    #pragma unroll
    for (int nf = 0; nf < 2; ++nf){
      #pragma unroll
      for (int r = 0; r < 8; ++r) acc[mf][nf][r] = 0.f;
    }
  }

  for (int kcH = 0; kcH < 2; ++kcH){
    // ---- (re)load 5 halo'd input rows, ci range [kcH*64, kcH*64+64), into LDS ----
    __syncthreads();   // all waves done with previous tile
    for (int idx = tid; idx < 5 * 132 * 8; idx += 256){
      int r   = idx / (132 * 8);
      int rem = idx - r * (132 * 8);
      int xp  = rem >> 3;          // padded x coordinate 0..131
      int q   = rem & 7;           // uint4 chunk within 64 ci halves
      int yy  = y + r - 2;
      int xs  = xp - 2;
      uint4 v = {0u, 0u, 0u, 0u};
      if ((unsigned)yy < 128u && (unsigned)xs < 128u)
        v = *(const uint4*)(Xt + ((((size_t)(b * 128 + yy)) * 128 + xs) * 128
                                  + kcH * 64 + q * 8));
      *(uint4*)(sX + ((size_t)r * XROW + xp * XPITCH + q * 8)) = v;
    }
    __syncthreads();

    for (int ky = 0; ky < 5; ++ky){
      #pragma unroll
      for (int kc2 = 0; kc2 < 2; ++kc2){
        int kc = kcH * 2 + kc2;
        #pragma unroll
        for (int kx = 0; kx < 5; ++kx){
          // B fragments straight from L2 (gW already in B-frag lane layout)
          const _Float16* wbase = gW + (((size_t)((ky * 5 + kx) * 4 + kc)) * 4096)
                                     + (nw * 2) * 512 + lane * 16;
          v16h b0, b1;
          ((uint4*)&b0)[0] = ((const uint4*)wbase)[0];
          ((uint4*)&b0)[1] = ((const uint4*)wbase)[1];
          ((uint4*)&b1)[0] = ((const uint4*)(wbase + 512))[0];
          ((uint4*)&b1)[1] = ((const uint4*)(wbase + 512))[1];
          // A fragments from LDS, batched
          v16h a[4];
          #pragma unroll
          for (int mf = 0; mf < 4; ++mf){
            int xp = x0w + mf * 16 + lm + kx;
            const _Float16* ap = sX + ((size_t)ky * XROW + xp * XPITCH
                                       + kc2 * 32 + kh * 16);
            ((uint4*)&a[mf])[0] = ((const uint4*)ap)[0];     // K = kh*16 + 0..7
            ((uint4*)&a[mf])[1] = *(const uint4*)(ap + 8);   // K = kh*16 + 8..15
          }
          #pragma unroll
          for (int mf = 0; mf < 4; ++mf){
            acc[mf][0] = __builtin_amdgcn_wmma_f32_16x16x32_f16(
                false, a[mf], false, b0, (short)0, acc[mf][0], false, false);
            acc[mf][1] = __builtin_amdgcn_wmma_f32_16x16x32_f16(
                false, a[mf], false, b1, (short)0, acc[mf][1], false, false);
          }
        }
      }
    }
  }

  // ---- transpose results through LDS for coalesced stores ----
  __syncthreads();
  float* sO = (float*)smem;   // 128 co x 132 (padded) floats = 67.6KB, reuses tile region
  #pragma unroll
  for (int mf = 0; mf < 4; ++mf){
    #pragma unroll
    for (int nf = 0; nf < 2; ++nf){
      #pragma unroll
      for (int r = 0; r < 8; ++r){
        int x  = x0w + mf * 16 + kh * 8 + r;     // D-frag: M = r + (lane/16)*8
        int co = nw * 32 + nf * 16 + lm;         // D-frag: N = lane%16
        sO[co * 132 + x] = acc[mf][nf][r];
      }
    }
  }
  __syncthreads();
  for (int i = tid; i < 4096; i += 256){
    int co = i >> 5, xq = i & 31;
    float4 v = *(float4*)(sO + co * 132 + xq * 4);
    int o = co & 15, nh = co >> 4;
    size_t off = ((((size_t)b * 16 + o) * 8 + nh) * 128 + y) * 128 + xq * 4;
    *(float4*)(out + off) = v;
  }
}

// ---------------- launcher ----------------
extern "C" void kernel_launch(void* const* d_in, const int* in_sizes, int n_in,
                              void* d_out, int out_size, void* d_ws, size_t ws_size,
                              hipStream_t stream){
  const float* X   = (const float*)d_in[0];   // (8,16,8,128,128) f32
  const float* wts = (const float*)d_in[1];   // (16,200,16) f32
  float* out = (float*)d_out;                 // (8,16,8,128,128) f32

  _Float16* Xt = (_Float16*)d_ws;                                   // 33,554,432 B
  _Float16* gW = (_Float16*)((char*)d_ws + (size_t)33554432);       //    819,200 B

  (void)hipFuncSetAttribute((const void*)k_conv,
                            hipFuncAttributeMaxDynamicSharedMemorySize, SMEM_BYTES);

  k_xpose  <<<dim3(4, 128, 8), 256, 0, stream>>>(X, Xt);
  k_weights<<<dim3(128),       256, 0, stream>>>(wts, gW);
  k_conv   <<<dim3(128, 8),    256, SMEM_BYTES, stream>>>(Xt, gW, out);
}
